// STTLayer_85031762526476
// MI455X (gfx1250) — matrix operand order repaired and checked by reading, not compiled, not run on hardware
//
#include <hip/hip_runtime.h>
#include <math.h>

// ---------------------------------------------------------------------------
// CDNA5 (gfx1250) STT layer. wave32, WMMA bf16 16x16x32 for all matmuls.
// GEMM K-loops are ping-pong double-buffered (2x manual unroll, K % 64 == 0):
// fragment loads for K-step k+32 fly under the WMMAs for K-step k, with no
// buffer-rotation register copies.
// ---------------------------------------------------------------------------

typedef __bf16 bf16_t;
typedef __attribute__((ext_vector_type(16))) __bf16 v16bf;
typedef __attribute__((ext_vector_type(8)))  __bf16 bf16x8;
typedef __attribute__((ext_vector_type(8)))  float  v8f;

#define WMMA_BF16(a, b, c) \
  __builtin_amdgcn_wmma_f32_16x16x32_bf16(false, (a), false, (b), (short)0, (c), false, false)

// A-fragment (16x32 bf16, MxK): lane r=lane&15 holds row r; lanes 0-15 take
// K = [0..7]+[16..23], lanes 16-31 take K = [8..15]+[24..31]. Contiguous 16B x2.
__device__ __forceinline__ v16bf load_afrag(const bf16_t* rowk0, int half) {
  v16bf a;
  ((bf16x8*)&a)[0] = *(const bf16x8*)(rowk0 + half * 8);
  ((bf16x8*)&a)[1] = *(const bf16x8*)(rowk0 + 16 + half * 8);
  return a;
}
// B-fragment (32x16 bf16, KxN) read from a [N,K] row-major array: lane r holds
// column N=r; lanes 0-15 take K = [0..7]+[16..23], lanes 16-31 take K = [8..15]+[24..31].
__device__ __forceinline__ v16bf load_bfrag(const bf16_t* rowk0, int half) {
  v16bf b;
  const bf16_t* p = rowk0 + half * 8;
  ((bf16x8*)&b)[0] = *(const bf16x8*)(p);
  ((bf16x8*)&b)[1] = *(const bf16x8*)(p + 16);
  return b;
}

// --------------------------- weight prep -----------------------------------
// src: fp32 [Kd, Nd] row-major  ->  dst: bf16 [Nd, Kd] row-major (transposed)
__global__ __launch_bounds__(256) void transpose_bf16_kernel(
    const float* __restrict__ src, bf16_t* __restrict__ dst, int Kd, int Nd) {
  __shared__ float tile[32][33];
  int bx = blockIdx.x * 32;            // N base
  int by = blockIdx.y * 32;            // K base
  int tx = threadIdx.x & 31, ty = threadIdx.x >> 5;   // 32 x 8
#pragma unroll
  for (int i = 0; i < 32; i += 8)
    tile[ty + i][tx] = src[(size_t)(by + ty + i) * Nd + bx + tx];
  __syncthreads();
#pragma unroll
  for (int i = 0; i < 32; i += 8)
    dst[(size_t)(bx + ty + i) * Kd + by + tx] = (bf16_t)tile[tx][ty + i];
}

// --------------------------- GEMM (bf16 WMMA) -------------------------------
// C[M,N] (f32) = A[M,K] (bf16 row-major) x W (given as Wt[N,K] bf16 row-major)
// block: 256 thr = 8 waves; wave tile 16x64; WG tile 64x128. K % 64 == 0.
__global__ __launch_bounds__(256) void gemm_bf16_kernel(
    const bf16_t* __restrict__ A, const bf16_t* __restrict__ Wt,
    float* __restrict__ C, int M, int N, int K) {
  int lane = threadIdx.x & 31, w = threadIdx.x >> 5;
  int half = lane >> 4, r = lane & 15;
  int arow = blockIdx.x * 64 + (w >> 1) * 16;
  int ncol = blockIdx.y * 128 + (w & 1) * 64;
  v8f acc[4];
#pragma unroll
  for (int t = 0; t < 4; ++t)
#pragma unroll
    for (int i = 0; i < 8; ++i) acc[t][i] = 0.0f;
  const bf16_t* Arow = A + (size_t)(arow + r) * K;
  const bf16_t* Brow[4];
#pragma unroll
  for (int nt = 0; nt < 4; ++nt)
    Brow[nt] = Wt + (size_t)(ncol + nt * 16 + r) * K;

  // ping-pong fragment sets (no buffer-rotation moves)
  v16bf a0 = load_afrag(Arow, half);
  v16bf b0[4];
#pragma unroll
  for (int nt = 0; nt < 4; ++nt) b0[nt] = load_bfrag(Brow[nt], half);
  v16bf a1, b1[4];

  int k0 = 0;
  for (; k0 + 64 < K; k0 += 64) {
    a1 = load_afrag(Arow + k0 + 32, half);
#pragma unroll
    for (int nt = 0; nt < 4; ++nt) b1[nt] = load_bfrag(Brow[nt] + k0 + 32, half);
#pragma unroll
    for (int nt = 0; nt < 4; ++nt) acc[nt] = WMMA_BF16(a0, b0[nt], acc[nt]);

    a0 = load_afrag(Arow + k0 + 64, half);
#pragma unroll
    for (int nt = 0; nt < 4; ++nt) b0[nt] = load_bfrag(Brow[nt] + k0 + 64, half);
#pragma unroll
    for (int nt = 0; nt < 4; ++nt) acc[nt] = WMMA_BF16(a1, b1[nt], acc[nt]);
  }
  // tail: exactly 64 K-elements left
  a1 = load_afrag(Arow + k0 + 32, half);
#pragma unroll
  for (int nt = 0; nt < 4; ++nt) b1[nt] = load_bfrag(Brow[nt] + k0 + 32, half);
#pragma unroll
  for (int nt = 0; nt < 4; ++nt) acc[nt] = WMMA_BF16(a0, b0[nt], acc[nt]);
#pragma unroll
  for (int nt = 0; nt < 4; ++nt) acc[nt] = WMMA_BF16(a1, b1[nt], acc[nt]);

#pragma unroll
  for (int nt = 0; nt < 4; ++nt)
#pragma unroll
    for (int i = 0; i < 8; ++i)
      C[(size_t)(arow + half * 8 + i) * N + ncol + nt * 16 + r] = acc[nt][i];
}

// Fused gate/up GEMM + SiLU epilogue: out bf16[M,N] = silu(A*Wg) * (A*Wu)
__global__ __launch_bounds__(256) void gemm_gateup_silu_kernel(
    const bf16_t* __restrict__ A, const bf16_t* __restrict__ WgT,
    const bf16_t* __restrict__ WuT, bf16_t* __restrict__ out,
    int M, int N, int K) {
  int lane = threadIdx.x & 31, w = threadIdx.x >> 5;
  int half = lane >> 4, r = lane & 15;
  int arow = blockIdx.x * 64 + (w >> 1) * 16;
  int ncol = blockIdx.y * 128 + (w & 1) * 64;
  v8f accg[4], accu[4];
#pragma unroll
  for (int t = 0; t < 4; ++t)
#pragma unroll
    for (int i = 0; i < 8; ++i) { accg[t][i] = 0.0f; accu[t][i] = 0.0f; }
  const bf16_t* Arow = A + (size_t)(arow + r) * K;
  const bf16_t* Bg[4]; const bf16_t* Bu[4];
#pragma unroll
  for (int nt = 0; nt < 4; ++nt) {
    size_t roff = (size_t)(ncol + nt * 16 + r) * K;
    Bg[nt] = WgT + roff; Bu[nt] = WuT + roff;
  }
  v16bf a0 = load_afrag(Arow, half);
  v16bf g0[4], u0[4];
#pragma unroll
  for (int nt = 0; nt < 4; ++nt) {
    g0[nt] = load_bfrag(Bg[nt], half);
    u0[nt] = load_bfrag(Bu[nt], half);
  }
  v16bf a1, g1[4], u1[4];

  int k0 = 0;
  for (; k0 + 64 < K; k0 += 64) {
    a1 = load_afrag(Arow + k0 + 32, half);
#pragma unroll
    for (int nt = 0; nt < 4; ++nt) {
      g1[nt] = load_bfrag(Bg[nt] + k0 + 32, half);
      u1[nt] = load_bfrag(Bu[nt] + k0 + 32, half);
    }
#pragma unroll
    for (int nt = 0; nt < 4; ++nt) {
      accg[nt] = WMMA_BF16(a0, g0[nt], accg[nt]);
      accu[nt] = WMMA_BF16(a0, u0[nt], accu[nt]);
    }
    a0 = load_afrag(Arow + k0 + 64, half);
#pragma unroll
    for (int nt = 0; nt < 4; ++nt) {
      g0[nt] = load_bfrag(Bg[nt] + k0 + 64, half);
      u0[nt] = load_bfrag(Bu[nt] + k0 + 64, half);
    }
#pragma unroll
    for (int nt = 0; nt < 4; ++nt) {
      accg[nt] = WMMA_BF16(a1, g1[nt], accg[nt]);
      accu[nt] = WMMA_BF16(a1, u1[nt], accu[nt]);
    }
  }
  a1 = load_afrag(Arow + k0 + 32, half);
#pragma unroll
  for (int nt = 0; nt < 4; ++nt) {
    g1[nt] = load_bfrag(Bg[nt] + k0 + 32, half);
    u1[nt] = load_bfrag(Bu[nt] + k0 + 32, half);
  }
#pragma unroll
  for (int nt = 0; nt < 4; ++nt) {
    accg[nt] = WMMA_BF16(a0, g0[nt], accg[nt]);
    accu[nt] = WMMA_BF16(a0, u0[nt], accu[nt]);
  }
#pragma unroll
  for (int nt = 0; nt < 4; ++nt) {
    accg[nt] = WMMA_BF16(a1, g1[nt], accg[nt]);
    accu[nt] = WMMA_BF16(a1, u1[nt], accu[nt]);
  }

#pragma unroll
  for (int nt = 0; nt < 4; ++nt)
#pragma unroll
    for (int i = 0; i < 8; ++i) {
      float g = accg[nt][i], u = accu[nt][i];
      float s = g / (1.0f + __expf(-g));
      out[(size_t)(arow + half * 8 + i) * N + ncol + nt * 16 + r] = (bf16_t)(s * u);
    }
}

// --------------------------- RMSNorm ---------------------------------------
__global__ __launch_bounds__(256) void rmsnorm_bf16_kernel(
    const float* __restrict__ x, const float* __restrict__ wt,
    bf16_t* __restrict__ y, int D) {
  int row = blockIdx.x;
  const float* xr = x + (size_t)row * D;
  __shared__ float red[256];
  float s = 0.0f;
  for (int i = threadIdx.x; i < D; i += 256) { float v = xr[i]; s += v * v; }
  red[threadIdx.x] = s; __syncthreads();
  for (int off = 128; off > 0; off >>= 1) {
    if (threadIdx.x < off) red[threadIdx.x] += red[threadIdx.x + off];
    __syncthreads();
  }
  float inv = rsqrtf(red[0] / (float)D + 1e-6f);
  for (int i = threadIdx.x; i < D; i += 256)
    y[(size_t)row * D + i] = (bf16_t)(xr[i] * inv * wt[i]);
}

// ---------------- RoPE + attention layout prep ------------------------------
// qf,kf,vf: f32 [B*Tl, 1024] (cols = h*64+d).  Outputs:
//   Q,K: bf16 [B,16,Tl,64]; VT: bf16 [B,16,64,Tl]. pos==null -> pos=t.
__global__ void qkv_prep_kernel(const float* __restrict__ qf,
                                const float* __restrict__ kf,
                                const float* __restrict__ vf,
                                const int* __restrict__ pos,
                                bf16_t* __restrict__ Q, bf16_t* __restrict__ K,
                                bf16_t* __restrict__ VT, int B, int Tl) {
  size_t idx = (size_t)blockIdx.x * blockDim.x + threadIdx.x;
  size_t total = (size_t)B * Tl * 16 * 32;
  if (idx >= total) return;
  int i = idx & 31; size_t tmp = idx >> 5;
  int h = (int)(tmp & 15); tmp >>= 4;
  int t = (int)(tmp % Tl); int b = (int)(tmp / Tl);
  int p = pos ? pos[b * Tl + t] : t;
  float inv_freq = __powf(10000.0f, -(float)(2 * i) / 64.0f);
  float ang = (float)p * inv_freq;
  float c = cosf(ang), s = sinf(ang);
  size_t base = ((size_t)b * Tl + t) * 1024 + h * 64;
  float q0 = qf[base + i], q1 = qf[base + i + 32];
  float k0 = kf[base + i], k1 = kf[base + i + 32];
  float v0 = vf[base + i], v1 = vf[base + i + 32];
  size_t qb = (((size_t)(b * 16 + h)) * Tl + t) * 64;
  Q[qb + i]      = (bf16_t)(q0 * c - q1 * s);
  Q[qb + i + 32] = (bf16_t)(q1 * c + q0 * s);
  K[qb + i]      = (bf16_t)(k0 * c - k1 * s);
  K[qb + i + 32] = (bf16_t)(k1 * c + k0 * s);
  size_t vtb = ((size_t)(b * 16 + h)) * 64;
  VT[(vtb + i) * Tl + t]      = (bf16_t)v0;
  VT[(vtb + i + 32) * Tl + t] = (bf16_t)v1;
}

// ---------------- flash attention (one wave per 16-row Q tile) --------------
// O: f32 [B, Tl, 16, 64] (== [B*Tl, 1024]). Index-causal mask (valid for both
// the dense pass and the compacted pass since top-k indices are sorted asc).
__global__ __launch_bounds__(256) void flash_attn_kernel(
    const bf16_t* __restrict__ Q, const bf16_t* __restrict__ K,
    const bf16_t* __restrict__ VT, float* __restrict__ O, int B, int Tl) {
  __shared__ bf16_t pbuf[8][16 * 32];
  int lane = threadIdx.x & 31, w = threadIdx.x >> 5;
  int half = lane >> 4, r = lane & 15;
  int gw = blockIdx.x * 8 + w;
  int nqt = Tl >> 4;
  int qt = gw % nqt; int h = (gw / nqt) & 15; int b = gw / (nqt * 16);
  int qrow0 = qt * 16;
  const bf16_t* Qbase = Q + (((size_t)(b * 16 + h)) * Tl + qrow0 + r) * 64;
  v16bf qa0 = load_afrag(Qbase, half);        // head-dims 0..31
  v16bf qa1 = load_afrag(Qbase + 32, half);   // head-dims 32..63
  v8f acc[4];
#pragma unroll
  for (int t = 0; t < 4; ++t)
#pragma unroll
    for (int i = 0; i < 8; ++i) acc[t][i] = 0.0f;
  float mrun[8], lrun[8], alpha[8];
#pragma unroll
  for (int i = 0; i < 8; ++i) { mrun[i] = -3.0e38f; lrun[i] = 0.0f; }

  int kend = qrow0 + 16;
  for (int kc = 0; kc < kend; kc += 32) {
    v8f s0, s1;
#pragma unroll
    for (int i = 0; i < 8; ++i) { s0[i] = 0.0f; s1[i] = 0.0f; }
    const bf16_t* K0 = K + (((size_t)(b * 16 + h)) * Tl + kc + r) * 64;
    s0 = WMMA_BF16(qa0, load_bfrag(K0, half), s0);
    s0 = WMMA_BF16(qa1, load_bfrag(K0 + 32, half), s0);
    const bf16_t* K1 = K0 + 16 * 64;
    s1 = WMMA_BF16(qa0, load_bfrag(K1, half), s1);
    s1 = WMMA_BF16(qa1, load_bfrag(K1 + 32, half), s1);
#pragma unroll
    for (int i = 0; i < 8; ++i) {
      int row = qrow0 + half * 8 + i;
      float v0 = (kc + r      <= row) ? s0[i] * 0.125f : -3.0e38f;
      float v1 = (kc + 16 + r <= row) ? s1[i] * 0.125f : -3.0e38f;
      float mx = fmaxf(v0, v1);
#pragma unroll
      for (int d = 1; d < 16; d <<= 1) mx = fmaxf(mx, __shfl_xor(mx, d, 32));
      float mnew = fmaxf(mrun[i], mx);
      float a = __expf(mrun[i] - mnew);
      float p0 = __expf(v0 - mnew), p1 = __expf(v1 - mnew);
      float ps = p0 + p1;
#pragma unroll
      for (int d = 1; d < 16; d <<= 1) ps += __shfl_xor(ps, d, 32);
      lrun[i] = lrun[i] * a + ps;
      mrun[i] = mnew; alpha[i] = a;
      pbuf[w][(half * 8 + i) * 32 + r] = (bf16_t)p0;
      pbuf[w][(half * 8 + i) * 32 + 16 + r] = (bf16_t)p1;
    }
#pragma unroll
    for (int nt = 0; nt < 4; ++nt)
#pragma unroll
      for (int i = 0; i < 8; ++i) acc[nt][i] *= alpha[i];
    // P (16x32) re-layout via per-wave LDS -> A-fragment (same-wave DS order)
    v16bf pa = load_afrag(&pbuf[w][r * 32], half);
#pragma unroll
    for (int nt = 0; nt < 4; ++nt) {
      const bf16_t* Vrow = VT + (((size_t)(b * 16 + h)) * 64 + nt * 16 + r) * Tl + kc;
      acc[nt] = WMMA_BF16(pa, load_bfrag(Vrow, half), acc[nt]);
    }
  }
#pragma unroll
  for (int nt = 0; nt < 4; ++nt)
#pragma unroll
    for (int i = 0; i < 8; ++i) {
      int row = qrow0 + half * 8 + i;
      O[(((size_t)b * Tl + row) * 16 + h) * 64 + nt * 16 + r] = acc[nt][i] / lrun[i];
    }
}

// ---------------- elementwise helpers ---------------------------------------
__global__ void f2bf_kernel(const float* __restrict__ a, bf16_t* __restrict__ y, size_t n) {
  size_t i = (size_t)blockIdx.x * blockDim.x + threadIdx.x;
  if (i < n) y[i] = (bf16_t)a[i];
}
__global__ void add_kernel(const float* __restrict__ a, const float* __restrict__ b,
                           float* __restrict__ c, size_t n) {
  size_t i = (size_t)blockIdx.x * blockDim.x + threadIdx.x;
  if (i < n) c[i] = a[i] + b[i];
}
__global__ void copy_kernel(const float* __restrict__ a, float* __restrict__ c, size_t n) {
  size_t i = (size_t)blockIdx.x * blockDim.x + threadIdx.x;
  if (i < n) c[i] = a[i];
}
__global__ void shift_kernel(const float* __restrict__ p, float* __restrict__ prev,
                             int B, int T, int D) {
  size_t i = (size_t)blockIdx.x * blockDim.x + threadIdx.x;
  size_t n = (size_t)B * T * D;
  if (i >= n) return;
  int d = (int)(i % D); size_t row = i / D;
  int t = (int)(row % T); int b = (int)(row / T);
  prev[i] = (t > 0) ? p[(((size_t)b * T) + t - 1) * D + d] : 0.0f;
}

// ---------------- surprise gate ---------------------------------------------
__global__ __launch_bounds__(256) void gate_kernel(
    const float* __restrict__ proc, const float* __restrict__ hid,
    const float* __restrict__ pred, const float* __restrict__ bce,
    const float* __restrict__ bcu, float* __restrict__ g, int D) {
  int row = blockIdx.x;
  __shared__ float r1[256], r2[256];
  float s1 = 0.0f, s2 = 0.0f;
  size_t base = (size_t)row * D;
  for (int i = threadIdx.x; i < D; i += 256) {
    float ar = proc[base + i] - hid[base + i];
    float df = ar - pred[base + i];
    s1 += ar * ar; s2 += df * df;
  }
  r1[threadIdx.x] = s1; r2[threadIdx.x] = s2; __syncthreads();
  for (int off = 128; off > 0; off >>= 1) {
    if (threadIdx.x < off) { r1[threadIdx.x] += r1[threadIdx.x + off];
                             r2[threadIdx.x] += r2[threadIdx.x + off]; }
    __syncthreads();
  }
  if (threadIdx.x == 0) {
    float dst = r1[0] / (float)D, dch = r2[0] / (float)D;
    g[row] = 1.0f / (1.0f + __expf(-(bcu[0] * dch - bce[0] * dst)));
  }
}

// ---------------- per-batch top-k + index sort ------------------------------
__global__ __launch_bounds__(1024) void topk_kernel(
    const float* __restrict__ g, int* __restrict__ idx_out,
    float* __restrict__ val_out, int T, int Ksel) {
  int b = blockIdx.x;
  __shared__ float v[2048];
  __shared__ int ix[2048];
  for (int i = threadIdx.x; i < T; i += 1024) { v[i] = g[b * T + i]; ix[i] = i; }
  __syncthreads();
  // bitonic: descending by value (tie: lower index first)
  for (int k = 2; k <= T; k <<= 1)
    for (int j = k >> 1; j > 0; j >>= 1) {
      for (int i = threadIdx.x; i < T; i += 1024) {
        int l = i ^ j;
        if (l > i) {
          bool up = ((i & k) == 0);
          bool before = (v[i] > v[l]) || (v[i] == v[l] && ix[i] < ix[l]);
          bool sw = up ? !before : before;
          if (sw) { float tv = v[i]; v[i] = v[l]; v[l] = tv;
                    int ti = ix[i]; ix[i] = ix[l]; ix[l] = ti; }
        }
      }
      __syncthreads();
    }
  // re-sort the top-Ksel ascending by original index
  for (int k = 2; k <= Ksel; k <<= 1)
    for (int j = k >> 1; j > 0; j >>= 1) {
      for (int i = threadIdx.x; i < Ksel; i += 1024) {
        int l = i ^ j;
        if (l > i && l < Ksel) {
          bool up = ((i & k) == 0);
          bool before = ix[i] < ix[l];
          bool sw = up ? !before : before;
          if (sw) { float tv = v[i]; v[i] = v[l]; v[l] = tv;
                    int ti = ix[i]; ix[i] = ix[l]; ix[l] = ti; }
        }
      }
      __syncthreads();
    }
  for (int i = threadIdx.x; i < Ksel; i += 1024) {
    idx_out[b * Ksel + i] = ix[i];
    val_out[b * Ksel + i] = v[i];
  }
}

// ---------------- gather / scatter ------------------------------------------
__global__ void gather_kernel(const float* __restrict__ x, const int* __restrict__ idx,
                              float* __restrict__ sel, int T, int Ksel, int D, size_t n) {
  size_t i = (size_t)blockIdx.x * blockDim.x + threadIdx.x;
  if (i >= n) return;
  int d = (int)(i % D); size_t row = i / D;
  int b = (int)(row / Ksel);
  sel[i] = x[(((size_t)b * T) + idx[row]) * D + d];
}
__global__ void scatter_kernel(const float* __restrict__ sel,
                               const float* __restrict__ procsel,
                               const float* __restrict__ vals,
                               const int* __restrict__ idx,
                               float* __restrict__ out, int T, int Ksel, int D, size_t n) {
  size_t i = (size_t)blockIdx.x * blockDim.x + threadIdx.x;
  if (i >= n) return;
  int d = (int)(i % D); size_t row = i / D;
  int b = (int)(row / Ksel);
  float s = sel[i];
  out[(((size_t)b * T) + idx[row]) * D + d] = s + vals[row] * (procsel[i] - s);
}

// ===========================================================================
extern "C" void kernel_launch(void* const* d_in, const int* in_sizes, int n_in,
                              void* d_out, int out_size, void* d_ws, size_t ws_size,
                              hipStream_t stream) {
  const int B = 2, T = 2048, D = 1024, F = 4096, KSEL = 1024;
  const float* hid    = (const float*)d_in[0];
  const float* wq     = (const float*)d_in[1];
  const float* wk     = (const float*)d_in[2];
  const float* wv     = (const float*)d_in[3];
  const float* wo     = (const float*)d_in[4];
  const float* ln1    = (const float*)d_in[5];
  const float* ln2    = (const float*)d_in[6];
  const float* wg     = (const float*)d_in[7];
  const float* wu     = (const float*)d_in[8];
  const float* wd     = (const float*)d_in[9];
  const float* tnn    = (const float*)d_in[10];
  const float* tn_wg  = (const float*)d_in[11];
  const float* tn_wu  = (const float*)d_in[12];
  const float* tn_wd  = (const float*)d_in[13];
  const float* bce    = (const float*)d_in[14];
  const float* bcu    = (const float*)d_in[15];

  const int R1 = B * T;          // 4096 rows, dense pass
  const int R2 = B * KSEL;       // 2048 rows, compacted pass

  char* p = (char*)d_ws;
  auto alloc = [&](size_t bytes) -> void* {
    void* r = (void*)p; p += (bytes + 255) & ~(size_t)255; return r;
  };
  // bf16 transposed weights [N,K]
  bf16_t* wqT  = (bf16_t*)alloc((size_t)D * D * 2);
  bf16_t* wkT  = (bf16_t*)alloc((size_t)D * D * 2);
  bf16_t* wvT  = (bf16_t*)alloc((size_t)D * D * 2);
  bf16_t* woT  = (bf16_t*)alloc((size_t)D * D * 2);
  bf16_t* wgT  = (bf16_t*)alloc((size_t)D * F * 2);
  bf16_t* wuT  = (bf16_t*)alloc((size_t)D * F * 2);
  bf16_t* wdT  = (bf16_t*)alloc((size_t)F * D * 2);
  bf16_t* tngT = (bf16_t*)alloc((size_t)D * F * 2);
  bf16_t* tnuT = (bf16_t*)alloc((size_t)D * F * 2);
  bf16_t* tndT = (bf16_t*)alloc((size_t)F * D * 2);
  // scratch (sized for the dense pass, reused by the compacted pass)
  bf16_t* xn_bf = (bf16_t*)alloc((size_t)R1 * D * 2); // also attn_bf / hn_bf
  float*  qf    = (float*)alloc((size_t)R1 * D * 4);  // also attn-out / mlp / prev
  float*  kf    = (float*)alloc((size_t)R1 * D * 4);  // also attnproj / pred
  float*  vf    = (float*)alloc((size_t)R1 * D * 4);  // also h
  bf16_t* Qbf   = (bf16_t*)alloc((size_t)R1 * D * 2);
  bf16_t* Kbf   = (bf16_t*)alloc((size_t)R1 * D * 2);
  bf16_t* VTb   = (bf16_t*)alloc((size_t)R1 * D * 2);
  bf16_t* mbuf  = (bf16_t*)alloc((size_t)R1 * F * 2);
  float*  proc  = (float*)alloc((size_t)R1 * D * 4);
  float*  gcont = (float*)alloc((size_t)B * T * 4);
  int*    idxb  = (int*)alloc((size_t)B * KSEL * 4);
  float*  valsb = (float*)alloc((size_t)B * KSEL * 4);
  float*  sel   = (float*)alloc((size_t)R2 * D * 4);
  float*  psel  = (float*)alloc((size_t)R2 * D * 4);

  auto nblk = [](size_t n) { return (unsigned)((n + 255) / 256); };

  // --- weight convert + transpose (fp32 [K,N] -> bf16 [N,K]) ---
  auto xpose = [&](const float* src, bf16_t* dst, int Kd, int Nd) {
    dim3 g(Nd / 32, Kd / 32);
    transpose_bf16_kernel<<<g, 256, 0, stream>>>(src, dst, Kd, Nd);
  };
  xpose(wq, wqT, D, D);  xpose(wk, wkT, D, D);
  xpose(wv, wvT, D, D);  xpose(wo, woT, D, D);
  xpose(wg, wgT, D, F);  xpose(wu, wuT, D, F);  xpose(wd, wdT, F, D);
  xpose(tn_wg, tngT, D, F); xpose(tn_wu, tnuT, D, F); xpose(tn_wd, tndT, F, D);

  // --- one decoder block: x[B*Tl,D] (pos per row) -> out[B*Tl,D] ---
  auto run_block = [&](const float* x, const int* pos, float* out, int Tl) {
    int R = B * Tl;
    size_t nRD = (size_t)R * D;
    dim3 gDD(R / 64, D / 128);
    rmsnorm_bf16_kernel<<<R, 256, 0, stream>>>(x, ln1, xn_bf, D);
    gemm_bf16_kernel<<<gDD, 256, 0, stream>>>(xn_bf, wqT, qf, R, D, D);
    gemm_bf16_kernel<<<gDD, 256, 0, stream>>>(xn_bf, wkT, kf, R, D, D);
    gemm_bf16_kernel<<<gDD, 256, 0, stream>>>(xn_bf, wvT, vf, R, D, D);
    qkv_prep_kernel<<<nblk((size_t)R * 512), 256, 0, stream>>>(
        qf, kf, vf, pos, Qbf, Kbf, VTb, B, Tl);
    flash_attn_kernel<<<B * 16 * (Tl / 16) / 8, 256, 0, stream>>>(
        Qbf, Kbf, VTb, qf /*attn out*/, B, Tl);
    f2bf_kernel<<<nblk(nRD), 256, 0, stream>>>(qf, xn_bf, nRD);
    gemm_bf16_kernel<<<gDD, 256, 0, stream>>>(xn_bf, woT, kf /*attnproj*/, R, D, D);
    add_kernel<<<nblk(nRD), 256, 0, stream>>>(x, kf, vf /*h*/, nRD);
    rmsnorm_bf16_kernel<<<R, 256, 0, stream>>>(vf, ln2, xn_bf, D);
    dim3 gDF(R / 64, F / 128);
    gemm_gateup_silu_kernel<<<gDF, 256, 0, stream>>>(xn_bf, wgT, wuT, mbuf, R, F, D);
    gemm_bf16_kernel<<<gDD, 256, 0, stream>>>(mbuf, wdT, qf /*mlp*/, R, D, F);
    add_kernel<<<nblk(nRD), 256, 0, stream>>>(vf, qf, out, nRD);
  };

  // 1) dense pass
  run_block(hid, nullptr, proc, T);

  // 2) transition network on shifted output
  {
    size_t nRD = (size_t)R1 * D;
    shift_kernel<<<nblk(nRD), 256, 0, stream>>>(proc, qf /*prev*/, B, T, D);
    rmsnorm_bf16_kernel<<<R1, 256, 0, stream>>>(qf, tnn, xn_bf, D);
    dim3 gDF(R1 / 64, F / 128);
    gemm_gateup_silu_kernel<<<gDF, 256, 0, stream>>>(xn_bf, tngT, tnuT, mbuf, R1, F, D);
    dim3 gDD(R1 / 64, D / 128);
    gemm_bf16_kernel<<<gDD, 256, 0, stream>>>(mbuf, tndT, kf /*pred*/, R1, D, F);
    // 3) surprise gate
    gate_kernel<<<R1, 256, 0, stream>>>(proc, hid, kf, bce, bcu, gcont, D);
  }

  // 4) top-k (per batch) with ascending-index reorder
  topk_kernel<<<B, 1024, 0, stream>>>(gcont, idxb, valsb, T, KSEL);

  // 5) gather -> second block pass -> soft-gated scatter
  size_t nSel = (size_t)R2 * D;
  gather_kernel<<<nblk(nSel), 256, 0, stream>>>(hid, idxb, sel, T, KSEL, D, nSel);
  run_block(sel, idxb, psel, KSEL);

  size_t nAll = (size_t)B * T * D;
  copy_kernel<<<nblk(nAll), 256, 0, stream>>>(hid, (float*)d_out, nAll);
  scatter_kernel<<<nblk(nSel), 256, 0, stream>>>(sel, psel, valsb, idxb,
                                                 (float*)d_out, T, KSEL, D, nSel);
}
